// DecoderLSTM_74165495267344
// MI455X (gfx1250) — matrix operand order repaired and checked by reading, not compiled, not run on hardware
//
#include <hip/hip_runtime.h>
#include <hip/hip_bf16.h>
#include <math.h>

// ---------------------------------------------------------------------------
// Problem constants (from reference)
// ---------------------------------------------------------------------------
#define L_     2
#define S_     128
#define H_     512
#define B_     512
#define G4_    (4 * H_)          // 2048 gate width
#define KC_    (2 * H_)          // 1024 concat-K for [x|h] @ [Wih;Whh]^T

typedef __bf16 bf16;
typedef __attribute__((ext_vector_type(8)))  bf16  v8bf;
typedef __attribute__((ext_vector_type(16))) bf16  v16bf;
typedef __attribute__((ext_vector_type(8)))  float v8f;

static __device__ __forceinline__ bf16 f2bf(float f) {
    unsigned u = __builtin_bit_cast(unsigned, f);
    unsigned r = (u + 0x7FFFu + ((u >> 16) & 1u)) >> 16;
    unsigned short s = (unsigned short)r;
    return __builtin_bit_cast(bf16, s);
}
static __device__ __forceinline__ float bf2f(bf16 b) {
    unsigned short s = __builtin_bit_cast(unsigned short, b);
    unsigned u = ((unsigned)s) << 16;
    return __builtin_bit_cast(float, u);
}
static __device__ __forceinline__ float sigm(float x) {
    return 1.0f / (1.0f + __expf(-x));
}

// ---------------------------------------------------------------------------
// Generic bf16 GEMM:  out[M,N] = A[M,K] * W[N,K]^T + bias
//   A   : bf16 row-major, leading dim lda
//   W   : bf16 row-major [N,K] (so B-operand column n == W row n), ld ldw
//   mode 0: outF[m*ldc+n] = v + bias[n]          (fp32 out)
//   mode 1: outB[m*ldc+n] = bf16(relu(v+bias))   (bf16 out)
// Grid: (N/64, M/64), block 128 (4 waves); each wave does a 16x64 tile.
// A/B WMMA fragments use contiguous k = 16*hi + s (same order in both).
// ---------------------------------------------------------------------------
__global__ void gemm_bf16_wmma(const bf16* __restrict__ A, int lda,
                               const bf16* __restrict__ W, int ldw,
                               const float* __restrict__ bias,
                               float* __restrict__ outF,
                               bf16*  __restrict__ outB,
                               int ldc, int K, int mode) {
    const int wave = threadIdx.x >> 5;
    const int lane = threadIdx.x & 31;
    const int lh   = lane & 15;        // 0..15
    const int hi   = lane >> 4;        // 0 or 1

    const int m_base = blockIdx.y * 64 + wave * 16;
    const int n_base = blockIdx.x * 64;

    v8f acc[4];
#pragma unroll
    for (int j = 0; j < 4; ++j)
#pragma unroll
        for (int e = 0; e < 8; ++e) acc[j][e] = 0.0f;

    const bf16* __restrict__ arow = A + (size_t)(m_base + lh) * lda;

    const int ksteps = K >> 5;
    for (int kt = 0; kt < ksteps; ++kt) {
        const int kb = kt * 32;
        // A fragment: lanes 0-15 hold K=[+0..15], lanes 16-31 [+16..31] contiguous
        v8bf alo = *(const v8bf*)(arow + kb + hi * 16);
        v8bf ahi = *(const v8bf*)(arow + kb + hi * 16 + 8);
        v16bf afrag = __builtin_shufflevector(alo, ahi,
            0, 1, 2, 3, 4, 5, 6, 7, 8, 9, 10, 11, 12, 13, 14, 15);

#pragma unroll
        for (int j = 0; j < 4; ++j) {
            // B fragment: lane n holds column (n_base+16j+lh), K=[kb+hi*16 .. +15] contiguous
            const bf16* bptr = W + (size_t)(n_base + j * 16 + lh) * ldw + kb + hi * 16;
            v16bf bfrag = *(const v16bf*)bptr;
            acc[j] = __builtin_amdgcn_wmma_f32_16x16x32_bf16(
                false, afrag, false, bfrag, (short)0, acc[j], false, false);
        }
    }

    // Epilogue: D layout — VGPR r: lanes 0-15 -> (M=r, N=lane); lanes 16-31 -> (M=8+r)
#pragma unroll
    for (int j = 0; j < 4; ++j) {
        const int n = n_base + j * 16 + lh;
        const float bv = bias ? bias[n] : 0.0f;
#pragma unroll
        for (int r = 0; r < 8; ++r) {
            const int m = m_base + hi * 8 + r;
            float v = acc[j][r] + bv;
            if (mode == 1) {
                v = v > 0.0f ? v : 0.0f;
                outB[(size_t)m * ldc + n] = f2bf(v);
            } else {
                outF[(size_t)m * ldc + n] = v;
            }
        }
    }
}

// ---------------------------------------------------------------------------
// LSTM cell elementwise. gates fp32 [B, 2048] in PyTorch order (i,f,g,o).
// layer0: hdst0 = A0+512 (h0 slot), hdst1 = A1 (x slot of layer1), h0src=null
// layer1: hdst0 = A1+512 (h1 slot), hdst1 = A0 (x slot next step),
//         h0src = A1 (h0 bf16), seqdst = seq + t*H   (row stride S_*H_)
// ---------------------------------------------------------------------------
__global__ void lstm_cell(const float* __restrict__ gates,
                          float* __restrict__ c,
                          const bf16* __restrict__ h0src,
                          bf16* __restrict__ hdst0,
                          bf16* __restrict__ hdst1,
                          bf16* __restrict__ seqdst) {
    const int idx = blockIdx.x * blockDim.x + threadIdx.x;  // 0 .. B*H-1
    const int b = idx >> 9;
    const int j = idx & (H_ - 1);
    const float* g = gates + (size_t)b * G4_;
    const float gi = g[j];
    const float gf = g[H_ + j];
    const float gg = g[2 * H_ + j];
    const float go = g[3 * H_ + j];
    const float cv = c[idx];
    const float cn = sigm(gf) * cv + sigm(gi) * tanhf(gg);
    const float hn = sigm(go) * tanhf(cn);
    c[idx] = cn;
    const size_t arow = (size_t)b * KC_ + j;
    hdst0[arow] = f2bf(hn);
    if (h0src) {
        const float hs = bf2f(h0src[arow]) + hn;
        const bf16 hsb = f2bf(hs);
        hdst1[arow] = hsb;
        seqdst[(size_t)b * (S_ * H_) + j] = hsb;
    } else {
        hdst1[arow] = f2bf(hn);
    }
}

// ---------------------------------------------------------------------------
// Helpers: zero / conversions
// ---------------------------------------------------------------------------
__global__ void zero_u32(unsigned* __restrict__ p, int n) {
    int i = blockIdx.x * blockDim.x + threadIdx.x;
    if (i < n) p[i] = 0u;
}
__global__ void conv_f32_bf16(bf16* __restrict__ dst, const float* __restrict__ src, int n) {
    int i = blockIdx.x * blockDim.x + threadIdx.x;
    if (i < n) dst[i] = f2bf(src[i]);
}
// x [B,H] fp32 -> A0 x-slot (stride KC_)
__global__ void conv_x(bf16* __restrict__ A0, const float* __restrict__ x) {
    int i = blockIdx.x * blockDim.x + threadIdx.x;  // 0..B*H-1
    int b = i >> 9, j = i & (H_ - 1);
    A0[(size_t)b * KC_ + j] = f2bf(x[i]);
}
// Wcomb[l][n][k] = (k<H) ? W_ih[l][n][k] : W_hh[l][n][k-H]   (bf16, [L,4H,2H])
__global__ void conv_wcomb(bf16* __restrict__ Wc,
                           const float* __restrict__ Wih,
                           const float* __restrict__ Whh) {
    int i = blockIdx.x * blockDim.x + threadIdx.x;  // 0 .. L*G4*KC-1
    int l = i >> 21;                  // G4_*KC_ = 2^21
    int rem = i & ((1 << 21) - 1);
    int n = rem >> 10;                // KC_ = 1024
    int k = rem & (KC_ - 1);
    float v = (k < H_) ? Wih[((size_t)l * G4_ + n) * H_ + k]
                       : Whh[((size_t)l * G4_ + n) * H_ + (k - H_)];
    Wc[i] = f2bf(v);
}
__global__ void add_bias(float* __restrict__ bc,
                         const float* __restrict__ bih,
                         const float* __restrict__ bhh, int n) {
    int i = blockIdx.x * blockDim.x + threadIdx.x;
    if (i < n) bc[i] = bih[i] + bhh[i];
}

// ---------------------------------------------------------------------------
// Host launch
// ---------------------------------------------------------------------------
extern "C" void kernel_launch(void* const* d_in, const int* in_sizes, int n_in,
                              void* d_out, int out_size, void* d_ws, size_t ws_size,
                              hipStream_t stream) {
    const float* x     = (const float*)d_in[0];
    const float* W_ih  = (const float*)d_in[1];
    const float* W_hh  = (const float*)d_in[2];
    const float* b_ih  = (const float*)d_in[3];
    const float* b_hh  = (const float*)d_in[4];
    const float* fc1_w = (const float*)d_in[5];
    const float* fc1_b = (const float*)d_in[6];
    const float* fc2_w = (const float*)d_in[7];
    const float* fc2_b = (const float*)d_in[8];
    const float* fc3_w = (const float*)d_in[9];
    const float* fc3_b = (const float*)d_in[10];
    float* out = (float*)d_out;

    // Workspace layout (bytes)
    char* ws = (char*)d_ws;
    size_t off = 0;
    auto take = [&](size_t bytes) { char* p = ws + off; off += (bytes + 255) & ~(size_t)255; return p; };

    bf16*  Wcomb = (bf16*)take((size_t)L_ * G4_ * KC_ * 2);      // 8 MB
    float* biasC = (float*)take((size_t)L_ * G4_ * 4);           // 16 KB
    bf16*  fcw1  = (bf16*)take((size_t)H_ * H_ * 2);
    bf16*  fcw2  = (bf16*)take((size_t)H_ * H_ * 2);
    bf16*  fcw3  = (bf16*)take((size_t)H_ * H_ * 2);
    bf16*  A0    = (bf16*)take((size_t)B_ * KC_ * 2);            // [x | h0]
    bf16*  A1    = (bf16*)take((size_t)B_ * KC_ * 2);            // [h0 | h1]
    float* cbuf  = (float*)take((size_t)L_ * B_ * H_ * 4);       // c state
    float* gates = (float*)take((size_t)B_ * G4_ * 4);           // 4 MB
    bf16*  seq   = (bf16*)take((size_t)B_ * S_ * H_ * 2);        // 64 MB (reused as Y2)
    bf16*  Y1    = (bf16*)take((size_t)B_ * S_ * H_ * 2);        // 64 MB
    (void)ws_size; (void)in_sizes; (void)n_in; (void)out_size;

    // --- init: zero A0, A1, c (contiguous is not guaranteed; zero each) ---
    {
        int nA = B_ * KC_ / 2;                 // bf16 pairs as u32
        zero_u32<<<(nA + 255) / 256, 256, 0, stream>>>((unsigned*)A0, nA);
        zero_u32<<<(nA + 255) / 256, 256, 0, stream>>>((unsigned*)A1, nA);
        int nC = L_ * B_ * H_;
        zero_u32<<<(nC + 255) / 256, 256, 0, stream>>>((unsigned*)cbuf, nC);
    }
    // --- conversions ---
    conv_x<<<(B_ * H_) / 256, 256, 0, stream>>>(A0, x);
    {
        int n = L_ * G4_ * KC_;
        conv_wcomb<<<n / 256, 256, 0, stream>>>(Wcomb, W_ih, W_hh);
        add_bias<<<(L_ * G4_) / 256, 256, 0, stream>>>(biasC, b_ih, b_hh, L_ * G4_);
        int nf = H_ * H_;
        conv_f32_bf16<<<nf / 256, 256, 0, stream>>>(fcw1, fc1_w, nf);
        conv_f32_bf16<<<nf / 256, 256, 0, stream>>>(fcw2, fc2_w, nf);
        conv_f32_bf16<<<nf / 256, 256, 0, stream>>>(fcw3, fc3_w, nf);
    }

    // --- recurrence: 128 sequential steps, 2 layers ---
    dim3 gemmGridR(G4_ / 64, B_ / 64);   // (32, 8)
    dim3 gemmBlk(128);
    dim3 cellGrid((B_ * H_) / 256);
    for (int t = 0; t < S_; ++t) {
        // layer 0: gates = [x|h0] @ Wcomb0^T + biasC0
        gemm_bf16_wmma<<<gemmGridR, gemmBlk, 0, stream>>>(
            A0, KC_, Wcomb, KC_, biasC, gates, nullptr, G4_, KC_, 0);
        lstm_cell<<<cellGrid, 256, 0, stream>>>(
            gates, cbuf, nullptr, A0 + H_, A1, nullptr);
        // layer 1: gates = [h0|h1] @ Wcomb1^T + biasC1
        gemm_bf16_wmma<<<gemmGridR, gemmBlk, 0, stream>>>(
            A1, KC_, Wcomb + (size_t)G4_ * KC_, KC_, biasC + G4_,
            gates, nullptr, G4_, KC_, 0);
        lstm_cell<<<cellGrid, 256, 0, stream>>>(
            gates, cbuf + B_ * H_, A1, A1 + H_, A0, seq + (size_t)t * H_);
    }

    // --- MLP head over M = B*S = 65536 rows ---
    const int M = B_ * S_;
    dim3 gemmGridH(H_ / 64, M / 64);     // (8, 1024)
    // fc1: relu -> Y1 (bf16)
    gemm_bf16_wmma<<<gemmGridH, gemmBlk, 0, stream>>>(
        seq, H_, fcw1, H_, fc1_b, nullptr, Y1, H_, H_, 1);
    // fc2: relu -> seq (reuse as Y2, bf16)
    gemm_bf16_wmma<<<gemmGridH, gemmBlk, 0, stream>>>(
        Y1, H_, fcw2, H_, fc2_b, nullptr, seq, H_, H_, 1);
    // fc3: -> d_out (fp32)
    gemm_bf16_wmma<<<gemmGridH, gemmBlk, 0, stream>>>(
        seq, H_, fcw3, H_, fc3_b, out, nullptr, H_, H_, 0);
}
